// GRUD_33560874451504
// MI455X (gfx1250) — compile-verified, matrix operand order checked by reading, not run on hardware
//
#include <hip/hip_runtime.h>
#include <hip/hip_bf16.h>

// ---------------------------------------------------------------------------
// GRU-D on MI455X (gfx1250, wave32, WMMA). B=512, T=200, F=64, H=256.
//
// Phase 1 (parallel over all B*T rows, bf16 WMMA):
//   gx = exp(-relu(d@Wgx+bgx)), gh = exp(-relu(d@Wgh+bgh)),
//   xh = m*x + (1-m)*(gx*xl + (1-gx)*xm),
//   A{z,r,h} = xh@W{z,r,h}x + m@W{z,r,h}m + b{z,r,h}   (stored f32, (t,b,H))
//
// Phase 2 (ONE persistent kernel, no global sync):
//   batch rows are independent chains -> 32 blocks x 16-row batch tiles.
//   16 waves/block; wave w owns H-columns [16w,16w+16):
//     - B-fragments of Wzh/Wrh/Whh register-cached for all 200 steps
//     - h state: f32 in registers + bf16 tile in LDS (A operand)
//     - per step: decay -> publish h -> barrier -> z,r WMMA -> rh to LDS
//                 -> barrier -> htilde WMMA -> gated update
//   global traffic per step = streamed Az/Ar/Ah/gh rows, prefetched 1 step
//   ahead via global_prefetch_b8.
// ---------------------------------------------------------------------------

typedef __bf16 bf16_t;
typedef __attribute__((ext_vector_type(16))) __bf16 v16bf;
typedef __attribute__((ext_vector_type(8)))  float  v8f;

#define GB 512     // batch
#define GT 200     // time steps
#define GF 64      // features
#define GH 256     // hidden
#define TBROWS (GT * GB)   // 102400 rows, flattened (t, b)

// Load one 16x32 bf16 WMMA operand fragment from a row-major (rows, K) array.
// ISA layout (16-bit operand, wave32): lane L -> row = L%16, half = L/16;
// elements 0..7  = K[k0 + half*8 .. +7], elements 8..15 = K[k0+16+half*8 .. +7]
__device__ __forceinline__ v16bf load_frag(const bf16_t* base, int ldk,
                                           int row0, int k0) {
  const int lane = threadIdx.x & 31;
  const int m    = lane & 15;
  const int half = lane >> 4;
  const bf16_t* p = base + (size_t)(row0 + m) * ldk + (k0 + half * 8);
  union Frag { struct { uint4 lo, hi; } u; v16bf v; } fr;
  fr.u.lo = *reinterpret_cast<const uint4*>(p);
  fr.u.hi = *reinterpret_cast<const uint4*>(p + 16);
  return fr.v;
}

#define WMMA_BF16(acc, a, b)                                                   \
  __builtin_amdgcn_wmma_f32_16x16x32_bf16(false, (a), false, (b), (short)0,    \
                                          (acc), false, false)

__device__ __forceinline__ float sigmoidf_(float x) {
  return 1.f / (1.f + expf(-x));
}

// ---------------------------------------------------------------------------
// Pack Delta and Mask from (b, t, f) f32 into (t, b, f) bf16 rows.
// ---------------------------------------------------------------------------
__global__ void grud_pack_inputs(const float* __restrict__ Delta,
                                 const float* __restrict__ Mask,
                                 bf16_t* __restrict__ deltaP,
                                 bf16_t* __restrict__ maskP) {
  size_t idx = (size_t)blockIdx.x * blockDim.x + threadIdx.x;
  const size_t n = (size_t)TBROWS * GF;
  if (idx >= n) return;
  int f = (int)(idx & (GF - 1));
  size_t rb = idx >> 6;            // row = t*GB + b
  int b = (int)(rb & (GB - 1));
  int t = (int)(rb >> 9);
  size_t src = ((size_t)b * GT + t) * GF + f;
  deltaP[idx] = (bf16_t)Delta[src];
  maskP[idx]  = (bf16_t)Mask[src];
}

// ---------------------------------------------------------------------------
// Pack all weight matrices (K, N) f32 -> transposed (N, K) bf16.
// ---------------------------------------------------------------------------
struct PackDesc { const float* src; bf16_t* dst; int K; int N; };
struct PackArgs { PackDesc d[11]; };

__global__ void grud_pack_weights(PackArgs pa) {
  for (int i = 0; i < 11; ++i) {
    const int K = pa.d[i].K, N = pa.d[i].N, n = K * N;
    for (int idx = blockIdx.x * blockDim.x + threadIdx.x; idx < n;
         idx += gridDim.x * blockDim.x) {
      int k = idx % K;
      int nn = idx / K;
      pa.d[i].dst[idx] = (bf16_t)pa.d[i].src[(size_t)k * N + nn];  // dst[nn*K+k]
    }
  }
}

// ---------------------------------------------------------------------------
// Phase 1: one wave per 16-row tile of the flattened (t, b) row space.
// 232 WMMAs worth of work per wave.
// ---------------------------------------------------------------------------
__global__ void __launch_bounds__(128) grud_phase1(
    const float* __restrict__ X, const float* __restrict__ Xl,
    const float* __restrict__ Mask, const float* __restrict__ Xmean,
    const bf16_t* __restrict__ deltaP, const bf16_t* __restrict__ maskP,
    const bf16_t* __restrict__ WgxP, const bf16_t* __restrict__ WghP,
    const bf16_t* __restrict__ WzxP, const bf16_t* __restrict__ WzmP,
    const bf16_t* __restrict__ WrxP, const bf16_t* __restrict__ WrmP,
    const bf16_t* __restrict__ WhxP, const bf16_t* __restrict__ WhmP,
    const float* __restrict__ bgx, const float* __restrict__ bgh,
    const float* __restrict__ bz, const float* __restrict__ br,
    const float* __restrict__ bh,
    float* __restrict__ ghBuf, float* __restrict__ Az,
    float* __restrict__ Ar, float* __restrict__ Ah) {
  const int wv   = threadIdx.x >> 5;
  const int lane = threadIdx.x & 31;
  const int half = lane >> 4;
  const int nl   = lane & 15;
  const int tile = blockIdx.x * 4 + wv;       // 6400 tiles
  const int r0   = tile * 16;                 // global row = t*GB + b
  const int t    = r0 >> 9;                   // 512 rows per t; same t for tile

  __shared__ __align__(16) bf16_t xhTile[4][16 * GF];
  bf16_t* xh = &xhTile[wv][0];

  // ---- gx (B=WgxP 64x64) + elementwise xh, stored bf16 in LDS ----
  for (int nt = 0; nt < 4; ++nt) {
    v8f acc = {};
#pragma unroll
    for (int kk = 0; kk < 2; ++kk) {
      v16bf a = load_frag(deltaP, GF, r0, kk * 32);
      v16bf b = load_frag(WgxP, GF, nt * 16, kk * 32);
      acc = WMMA_BF16(acc, a, b);
    }
    const int f = nt * 16 + nl;
#pragma unroll
    for (int r = 0; r < 8; ++r) {
      const int lrow = r + 8 * half;
      const int row  = r0 + lrow;
      const int b_   = row & (GB - 1);
      const size_t src = ((size_t)b_ * GT + t) * GF + f;
      float gx = expf(-fmaxf(acc[r] + bgx[f], 0.f));
      float m  = Mask[src];
      float xv = X[src];
      float xm = Xmean[(size_t)t * GF + f];
      float xhv = m * xv + (1.f - m) * (gx * Xl[src] + (1.f - gx) * xm);
      xh[lrow * GF + f] = (bf16_t)xhv;
    }
  }
  __syncthreads();

  // ---- gh = exp(-relu(d@Wgh + bgh)), stored f32 (t,b,H) ----
  for (int nt = 0; nt < 16; ++nt) {
    v8f acc = {};
#pragma unroll
    for (int kk = 0; kk < 2; ++kk) {
      v16bf a = load_frag(deltaP, GF, r0, kk * 32);
      v16bf b = load_frag(WghP, GF, nt * 16, kk * 32);
      acc = WMMA_BF16(acc, a, b);
    }
    const int col = nt * 16 + nl;
#pragma unroll
    for (int r = 0; r < 8; ++r) {
      const int row = r0 + r + 8 * half;
      ghBuf[(size_t)row * GH + col] = expf(-fmaxf(acc[r] + bgh[col], 0.f));
    }
  }

  // ---- A{z,r,h} = xh@W*x + m@W*m + b*, stored f32 (t,b,H) ----
  const bf16_t* Wx[3]  = {WzxP, WrxP, WhxP};
  const bf16_t* Wm[3]  = {WzmP, WrmP, WhmP};
  const float*  bv[3]  = {bz, br, bh};
  float*        Out[3] = {Az, Ar, Ah};
#pragma unroll
  for (int g = 0; g < 3; ++g) {
    for (int nt = 0; nt < 16; ++nt) {
      v8f acc = {};
#pragma unroll
      for (int kk = 0; kk < 2; ++kk) {
        v16bf a = load_frag(xh, GF, 0, kk * 32);
        v16bf b = load_frag(Wx[g], GF, nt * 16, kk * 32);
        acc = WMMA_BF16(acc, a, b);
      }
#pragma unroll
      for (int kk = 0; kk < 2; ++kk) {
        v16bf a = load_frag(maskP, GF, r0, kk * 32);
        v16bf b = load_frag(Wm[g], GF, nt * 16, kk * 32);
        acc = WMMA_BF16(acc, a, b);
      }
      const int col = nt * 16 + nl;
#pragma unroll
      for (int r = 0; r < 8; ++r) {
        const int row = r0 + r + 8 * half;
        Out[g][(size_t)row * GH + col] = acc[r] + bv[g][col];
      }
    }
  }
}

// ---------------------------------------------------------------------------
// Phase 2: persistent recurrence. 32 blocks x 512 threads (16 waves).
// Block j owns batch rows [16j, 16j+16); wave w owns H-cols [16w, 16w+16).
// Weights register-cached; h in registers (f32) + LDS (bf16 A-operand).
// ---------------------------------------------------------------------------
__global__ void __launch_bounds__(512) grud_recurrence(
    const bf16_t* __restrict__ WzhP, const bf16_t* __restrict__ WrhP,
    const bf16_t* __restrict__ WhhP,
    const float* __restrict__ Az, const float* __restrict__ Ar,
    const float* __restrict__ Ah, const float* __restrict__ ghBuf,
    float* __restrict__ out) {
  const int w    = threadIdx.x >> 5;       // wave id 0..15 = N-tile
  const int lane = threadIdx.x & 31;
  const int half = lane >> 4;
  const int nl   = lane & 15;
  const int b0   = blockIdx.x * 16;        // batch tile base row

  __shared__ __align__(16) bf16_t hTile[16 * GH];    // 8 KB, decayed h (bf16)
  __shared__ __align__(16) bf16_t rhTile[16 * GH];   // 8 KB, r*h (bf16)

  // Register-cache all B fragments for the scan: 24 x v16bf = 192 VGPRs.
  v16bf Bz[8], Br[8], Bh[8];
#pragma unroll
  for (int kk = 0; kk < 8; ++kk) {
    Bz[kk] = load_frag(WzhP, GH, w * 16, kk * 32);
    Br[kk] = load_frag(WrhP, GH, w * 16, kk * 32);
    Bh[kk] = load_frag(WhhP, GH, w * 16, kk * 32);
  }

  const int col = w * 16 + nl;
  // f32 hidden state for the 8 (row, col) elements this lane owns (C-layout).
  float h[8];
#pragma unroll
  for (int r = 0; r < 8; ++r) h[r] = 0.f;

  // prefetch thread mapping: 512 threads cover 16 rows x 32 segments of 8 f32
  const int pfRow = threadIdx.x >> 5;
  const int pfSeg = (threadIdx.x & 31) * 8;

  for (int t = 0; t < GT; ++t) {
    // ---- prefetch next step's streamed epilogue data (global_prefetch_b8)
    if (t + 1 < GT) {
      const size_t pfi = ((size_t)(t + 1) * GB + b0 + pfRow) * GH + pfSeg;
      __builtin_prefetch(&Az[pfi], 0, 2);
      __builtin_prefetch(&Ar[pfi], 0, 2);
      __builtin_prefetch(&Ah[pfi], 0, 2);
      __builtin_prefetch(&ghBuf[pfi], 0, 2);
    }

    // ---- 1. decay h with gh[t], publish bf16 tile for A-operands
    float hd[8];
#pragma unroll
    for (int r = 0; r < 8; ++r) {
      const int brow = r + 8 * half;
      float g = ghBuf[((size_t)t * GB + b0 + brow) * GH + col];
      hd[r] = g * h[r];
      hTile[brow * GH + col] = (bf16_t)hd[r];
    }
    __syncthreads();

    // ---- 2. z and r GEMMs (two independent 8-deep WMMA chains)
    v8f za = {}, ra = {};
#pragma unroll
    for (int kk = 0; kk < 8; ++kk) {
      v16bf a = load_frag(hTile, GH, 0, kk * 32);
      za = WMMA_BF16(za, a, Bz[kk]);
      ra = WMMA_BF16(ra, a, Br[kk]);
    }
    float z[8];
#pragma unroll
    for (int r = 0; r < 8; ++r) {
      const int brow = r + 8 * half;
      const size_t gi = ((size_t)t * GB + b0 + brow) * GH + col;
      z[r] = sigmoidf_(Az[gi] + za[r]);
      float rr = sigmoidf_(Ar[gi] + ra[r]);
      rhTile[brow * GH + col] = (bf16_t)(rr * hd[r]);
    }
    __syncthreads();

    // ---- 3. candidate GEMM + gated update (h stays in registers)
    v8f ha = {};
#pragma unroll
    for (int kk = 0; kk < 8; ++kk) {
      v16bf a = load_frag(rhTile, GH, 0, kk * 32);
      ha = WMMA_BF16(ha, a, Bh[kk]);
    }
#pragma unroll
    for (int r = 0; r < 8; ++r) {
      const int brow = r + 8 * half;
      const size_t gi = ((size_t)t * GB + b0 + brow) * GH + col;
      float ht = tanhf(Ah[gi] + ha[r]);
      h[r] = (1.f - z[r]) * hd[r] + z[r] * ht;
    }
  }

  // ---- write final hidden state (B, H) f32
#pragma unroll
  for (int r = 0; r < 8; ++r) {
    out[(size_t)(b0 + r + 8 * half) * GH + col] = h[r];
  }
}

// ---------------------------------------------------------------------------
extern "C" void kernel_launch(void* const* d_in, const int* in_sizes, int n_in,
                              void* d_out, int out_size, void* d_ws,
                              size_t ws_size, hipStream_t stream) {
  const float* X     = (const float*)d_in[0];
  const float* Xl    = (const float*)d_in[1];
  const float* Mask  = (const float*)d_in[2];
  const float* Delta = (const float*)d_in[3];
  const float* Xmean = (const float*)d_in[4];
  const float* Wgx = (const float*)d_in[5];  const float* bgx = (const float*)d_in[6];
  const float* Wgh = (const float*)d_in[7];  const float* bgh = (const float*)d_in[8];
  const float* Wzx = (const float*)d_in[9];  const float* Wzh = (const float*)d_in[10];
  const float* Wzm = (const float*)d_in[11]; const float* bz  = (const float*)d_in[12];
  const float* Wrx = (const float*)d_in[13]; const float* Wrh = (const float*)d_in[14];
  const float* Wrm = (const float*)d_in[15]; const float* br  = (const float*)d_in[16];
  const float* Whx = (const float*)d_in[17]; const float* Whh = (const float*)d_in[18];
  const float* Whm = (const float*)d_in[19]; const float* bh  = (const float*)d_in[20];
  float* out = (float*)d_out;

  char* ws = (char*)d_ws;
  size_t off = 0;
  auto alloc = [&](size_t bytes) -> void* {
    void* p = ws + off;
    off += (bytes + 255) & ~(size_t)255;
    return p;
  };

  bf16_t* deltaP = (bf16_t*)alloc((size_t)TBROWS * GF * 2);
  bf16_t* maskP  = (bf16_t*)alloc((size_t)TBROWS * GF * 2);
  bf16_t* WgxP = (bf16_t*)alloc(64 * 64 * 2);
  bf16_t* WghP = (bf16_t*)alloc(256 * 64 * 2);
  bf16_t* WzxP = (bf16_t*)alloc(256 * 64 * 2);
  bf16_t* WzmP = (bf16_t*)alloc(256 * 64 * 2);
  bf16_t* WrxP = (bf16_t*)alloc(256 * 64 * 2);
  bf16_t* WrmP = (bf16_t*)alloc(256 * 64 * 2);
  bf16_t* WhxP = (bf16_t*)alloc(256 * 64 * 2);
  bf16_t* WhmP = (bf16_t*)alloc(256 * 64 * 2);
  bf16_t* WzhP = (bf16_t*)alloc(256 * 256 * 2);
  bf16_t* WrhP = (bf16_t*)alloc(256 * 256 * 2);
  bf16_t* WhhP = (bf16_t*)alloc(256 * 256 * 2);
  float* ghBuf = (float*)alloc((size_t)TBROWS * GH * 4);
  float* AzB   = (float*)alloc((size_t)TBROWS * GH * 4);
  float* ArB   = (float*)alloc((size_t)TBROWS * GH * 4);
  float* AhB   = (float*)alloc((size_t)TBROWS * GH * 4);
  (void)ws_size; (void)in_sizes; (void)n_in; (void)out_size;

  // Input repack: (b,t,f) f32 -> (t,b,f) bf16
  const int npack = TBROWS * GF;
  grud_pack_inputs<<<(npack + 255) / 256, 256, 0, stream>>>(Delta, Mask,
                                                            deltaP, maskP);
  // Weight repack: (K,N) f32 -> (N,K) bf16
  PackArgs pa = {{{Wgx, WgxP, 64, 64},   {Wgh, WghP, 64, 256},
                  {Wzx, WzxP, 64, 256},  {Wzm, WzmP, 64, 256},
                  {Wrx, WrxP, 64, 256},  {Wrm, WrmP, 64, 256},
                  {Whx, WhxP, 64, 256},  {Whm, WhmP, 64, 256},
                  {Wzh, WzhP, 256, 256}, {Wrh, WrhP, 256, 256},
                  {Whh, WhhP, 256, 256}}};
  grud_pack_weights<<<128, 256, 0, stream>>>(pa);

  // Phase 1: all input-side GEMMs (6400 tiles, 4 waves/block)
  grud_phase1<<<TBROWS / 16 / 4, 128, 0, stream>>>(
      X, Xl, Mask, Xmean, deltaP, maskP, WgxP, WghP, WzxP, WzmP, WrxP, WrmP,
      WhxP, WhmP, bgx, bgh, bz, br, bh, ghBuf, AzB, ArB, AhB);

  // Phase 2: one persistent kernel runs the entire 200-step recurrence
  grud_recurrence<<<GB / 16, 512, 0, stream>>>(WzhP, WrhP, WhhP, AzB, ArB,
                                               AhB, ghBuf, out);
}
